// Attention_56178172231784
// MI455X (gfx1250) — matrix operand order repaired and checked by reading, not compiled, not run on hardware
//
#include <hip/hip_runtime.h>
#include <hip/hip_bf16.h>

typedef __attribute__((ext_vector_type(16))) _Float16 v16h;
typedef __attribute__((ext_vector_type(8)))  _Float16 v8h;
typedef __attribute__((ext_vector_type(8)))  float    v8f;

#define DIM     512
#define HEADS   8
#define DH      32
#define INNER   256
#define NQKV    768
#define BATCH   2
#define SEQ     2048
#define BN      (BATCH * SEQ)   // 4096
#define QK_SCALE 0.1767766952966369f  // 32^-0.5

__device__ __forceinline__ v8f wmma_f16(v16h a, v16h b, v8f c) {
    // D = A(16x32 f16) * B(32x16 f16) + C(16x16 f32)
    return __builtin_amdgcn_wmma_f32_16x16x32_f16(
        /*neg_a=*/false, a, /*neg_b=*/false, b,
        /*c_mod=*/(short)0, c, /*reuse_a=*/false, /*reuse_b=*/false);
}

__device__ __forceinline__ v16h cat16(v8h lo, v8h hi) {
    return __builtin_shufflevector(lo, hi, 0,1,2,3,4,5,6,7,8,9,10,11,12,13,14,15);
}

// ---------------------------------------------------------------------------
// 1) LayerNorm (no bias) + cast to f16. One row per block, 512 threads.
// ---------------------------------------------------------------------------
__global__ __launch_bounds__(512) void ln_kernel(const float* __restrict__ x,
                                                 const float* __restrict__ g,
                                                 _Float16* __restrict__ xn) {
    __shared__ float red[2][16];
    const int row = blockIdx.x;
    const int t = threadIdx.x;
    const float v = x[(size_t)row * DIM + t];
    float s1 = v, s2 = v * v;
    #pragma unroll
    for (int m = 16; m; m >>= 1) {
        s1 += __shfl_xor(s1, m, 32);
        s2 += __shfl_xor(s2, m, 32);
    }
    if ((t & 31) == 0) { red[0][t >> 5] = s1; red[1][t >> 5] = s2; }
    __syncthreads();
    float S1 = 0.f, S2 = 0.f;
    #pragma unroll
    for (int i = 0; i < 16; ++i) { S1 += red[0][i]; S2 += red[1][i]; }
    const float mu  = S1 * (1.0f / DIM);
    const float var = S2 * (1.0f / DIM) - mu * mu;
    const float r   = rsqrtf(var + 1e-5f);
    xn[(size_t)row * DIM + t] = (_Float16)((v - mu) * r * g[t]);
}

// ---------------------------------------------------------------------------
// 2) Pack a row-major f32 [K][Nn] weight into per-lane WMMA B-operand tiles:
//    P[((kt*(Nn/16)+nt)*32 + lane)*16 + i] = W[kt*32 + (lane>>4)*16 + i][nt*16 + (lane&15)]
// ---------------------------------------------------------------------------
__global__ void pack_b(const float* __restrict__ W, _Float16* __restrict__ P,
                       int K, int Nn) {
    const int tid  = blockIdx.x * blockDim.x + threadIdx.x;
    const int lane = tid & 31;
    const int tile = tid >> 5;
    const int ntiles = (K / 32) * (Nn / 16);
    if (tile >= ntiles) return;
    const int nt = tile % (Nn / 16);
    const int kt = tile / (Nn / 16);
    const int col = nt * 16 + (lane & 15);
    const int kb  = kt * 32 + (lane >> 4) * 16;
    _Float16* dst = P + (size_t)tile * 512 + (size_t)lane * 16;
    #pragma unroll
    for (int i = 0; i < 16; ++i)
        dst[i] = (_Float16)W[(size_t)(kb + i) * Nn + col];
}

// ---------------------------------------------------------------------------
// 3) QKV GEMM: [4096x512] f16 @ packed [512x768] -> q (scaled, head-major),
//    k (head-major row-major), v (head-major TRANSPOSED [b][h][d][n]).
//    8 waves / block, each wave one 16x16 tile. Two independent accumulator
//    chains so consecutive WMMAs have no D->C RAW dependency.
// ---------------------------------------------------------------------------
__global__ __launch_bounds__(256) void qkv_gemm(const _Float16* __restrict__ xn,
                                                const _Float16* __restrict__ wp,
                                                _Float16* __restrict__ qd,
                                                _Float16* __restrict__ kd,
                                                _Float16* __restrict__ vtd) {
    const int lane = threadIdx.x & 31;
    const int wave = threadIdx.x >> 5;
    const int mt = blockIdx.x;
    const int nt = blockIdx.y * 8 + wave;        // 0..47
    const int row = mt * 16 + (lane & 15);
    const int kb  = (lane >> 4) * 16;
    v8f acc0 = {}, acc1 = {};
    #pragma unroll 2
    for (int kt = 0; kt < DIM / 32; kt += 2) {
        const _Float16* ap0 = xn + (size_t)row * DIM + kt * 32 + kb;
        const _Float16* ap1 = ap0 + 32;
        v16h a0 = cat16(*(const v8h*)ap0, *(const v8h*)(ap0 + 8));
        v16h a1 = cat16(*(const v8h*)ap1, *(const v8h*)(ap1 + 8));
        v16h b0 = *(const v16h*)(wp + (((size_t)kt * (NQKV / 16) + nt) * 32 + lane) * 16);
        v16h b1 = *(const v16h*)(wp + (((size_t)(kt + 1) * (NQKV / 16) + nt) * 32 + lane) * 16);
        acc0 = wmma_f16(a0, b0, acc0);
        acc1 = wmma_f16(a1, b1, acc1);
    }
    const int col = nt * 16 + (lane & 15);
    #pragma unroll
    for (int i = 0; i < 8; ++i) {
        const int r = mt * 16 + i + (lane >> 4) * 8;   // global token 0..4095
        const int bb = r >> 11, n = r & (SEQ - 1);
        const float v = acc0[i] + acc1[i];
        if (col < INNER) {                       // Q, pre-scaled
            const int h = col >> 5, d = col & 31;
            qd[(((size_t)bb * HEADS + h) * SEQ + n) * DH + d] = (_Float16)(v * QK_SCALE);
        } else if (col < 2 * INNER) {            // K, row-major per head
            const int c = col - INNER, h = c >> 5, d = c & 31;
            kd[(((size_t)bb * HEADS + h) * SEQ + n) * DH + d] = (_Float16)v;
        } else {                                 // V, transposed [d][n] per head
            const int c = col - 2 * INNER, h = c >> 5, d = c & 31;
            vtd[(((size_t)bb * HEADS + h) * DH + d) * SEQ + n] = (_Float16)v;
        }
    }
}

// ---------------------------------------------------------------------------
// 4) Fused attention: one (head, 16-query tile) per wave, BOTH batches per
//    wave so the 134 MB rel_pos_bias stream is read exactly once. Flash-style
//    online softmax, 32 keys per step, all matmuls on WMMA. The bias stream
//    (the roofline-dominant traffic) is prefetched 4 key-blocks ahead.
// ---------------------------------------------------------------------------
__global__ __launch_bounds__(128) void attn_kernel(const _Float16* __restrict__ q,
                                                   const _Float16* __restrict__ k,
                                                   const _Float16* __restrict__ vt,
                                                   const float* __restrict__ bias,
                                                   _Float16* __restrict__ o) {
    __shared__ _Float16 plds[4][2][16][32];      // per-wave, per-batch P tile
    const int lane = threadIdx.x & 31;
    const int wave = threadIdx.x >> 5;
    const int h = blockIdx.y;
    const int qbase = (blockIdx.x * 4 + wave) * 16;
    const int hi16 = lane >> 4;                  // 0/1 half-wave
    const int l15  = lane & 15;

    // Q A-fragments (fixed for the whole key loop), one per batch
    v16h aq[2];
    #pragma unroll
    for (int bb = 0; bb < 2; ++bb) {
        const _Float16* Qh = q + (((size_t)bb * HEADS + h) * SEQ + qbase + l15) * DH + hi16 * 16;
        aq[bb] = cat16(*(const v8h*)Qh, *(const v8h*)(Qh + 8));
    }

    float m[2][8], s[2][8];
    v8f o0[2] = {}, o1[2] = {};
    #pragma unroll
    for (int bb = 0; bb < 2; ++bb)
        #pragma unroll
        for (int i = 0; i < 8; ++i) { m[bb][i] = -3.0e38f; s[bb][i] = 0.f; }

    // bias row base for this lane's fragment rows (row i -> + i*SEQ)
    const float* bp = bias + ((size_t)h * SEQ + qbase + hi16 * 8) * SEQ;

    for (int j0 = 0; j0 < SEQ; j0 += 32) {
        // stream-prefetch the bias 4 key-blocks (128 floats) ahead into GL2
        #pragma unroll
        for (int i = 0; i < 8; ++i)
            __builtin_prefetch(bp + (size_t)i * SEQ + j0 + 128 + l15, 0, 0);
        // ---- bias: loaded ONCE, shared across both batches ----
        float bias0[8], bias1[8];
        #pragma unroll
        for (int i = 0; i < 8; ++i) {
            const float* r = bp + (size_t)i * SEQ + j0 + l15;
            bias0[i] = r[0];
            bias1[i] = r[16];
        }
        #pragma unroll
        for (int bb = 0; bb < 2; ++bb) {
            const _Float16* Kh  = k  + (((size_t)bb * HEADS + h) * SEQ) * DH;
            const _Float16* Vth = vt + (((size_t)bb * HEADS + h) * DH) * SEQ;

            // sim = Q @ K^T : row-major K is natively the WMMA B-operand (Dh==32==K)
            v16h bk0 = *(const v16h*)(Kh + (size_t)(j0 + l15) * DH + hi16 * 16);
            v16h bk1 = *(const v16h*)(Kh + (size_t)(j0 + 16 + l15) * DH + hi16 * 16);
            v8f z = {};
            v8f sim0 = wmma_f16(aq[bb], bk0, z);
            v8f sim1 = wmma_f16(aq[bb], bk1, z);

            float mn[8], alpha[8], p0[8], p1[8];
            #pragma unroll
            for (int i = 0; i < 8; ++i) { sim0[i] += bias0[i]; sim1[i] += bias1[i]; }
            // per-row max over the 32 columns (reduction stays inside a 16-lane half)
            #pragma unroll
            for (int i = 0; i < 8; ++i) {
                float mx = fmaxf(sim0[i], sim1[i]);
                #pragma unroll
                for (int msk = 8; msk; msk >>= 1) mx = fmaxf(mx, __shfl_xor(mx, msk, 32));
                mn[i] = fmaxf(m[bb][i], mx);
                alpha[i] = __expf(m[bb][i] - mn[i]);
                m[bb][i] = mn[i];
            }
            #pragma unroll
            for (int i = 0; i < 8; ++i) {
                p0[i] = __expf(sim0[i] - mn[i]);
                p1[i] = __expf(sim1[i] - mn[i]);
                float rs = p0[i] + p1[i];
                #pragma unroll
                for (int msk = 8; msk; msk >>= 1) rs += __shfl_xor(rs, msk, 32);
                s[bb][i] = s[bb][i] * alpha[i] + rs;
                o0[bb][i] *= alpha[i];
                o1[bb][i] *= alpha[i];
            }
            // C-fragment -> A-fragment layout conversion through LDS
            #pragma unroll
            for (int i = 0; i < 8; ++i) {
                plds[wave][bb][i + hi16 * 8][l15]      = (_Float16)p0[i];
                plds[wave][bb][i + hi16 * 8][16 + l15] = (_Float16)p1[i];
            }
            const _Float16* pr = &plds[wave][bb][l15][hi16 * 16];
            v16h ap = cat16(*(const v8h*)pr, *(const v8h*)(pr + 8));

            // o += P @ V : transposed V makes the B-operand loads contiguous
            v16h bv0 = *(const v16h*)(Vth + (size_t)l15 * SEQ + j0 + hi16 * 16);
            v16h bv1 = *(const v16h*)(Vth + (size_t)(16 + l15) * SEQ + j0 + hi16 * 16);
            o0[bb] = wmma_f16(ap, bv0, o0[bb]);
            o1[bb] = wmma_f16(ap, bv1, o1[bb]);
        }
    }
    // normalize and store o as [b][n][h*32+d] f16
    #pragma unroll
    for (int bb = 0; bb < 2; ++bb)
        #pragma unroll
        for (int i = 0; i < 8; ++i) {
            const int r = qbase + i + hi16 * 8;
            const float inv = 1.0f / s[bb][i];
            _Float16* dst = o + ((size_t)(bb * SEQ + r)) * INNER + h * DH;
            dst[l15]      = (_Float16)(o0[bb][i] * inv);
            dst[16 + l15] = (_Float16)(o1[bb][i] * inv);
        }
}

// ---------------------------------------------------------------------------
// 5) Output projection: [4096x256] f16 @ packed [256x512] + b_out -> f32 out
// ---------------------------------------------------------------------------
__global__ __launch_bounds__(256) void out_gemm(const _Float16* __restrict__ oin,
                                                const _Float16* __restrict__ wp,
                                                const float* __restrict__ bo,
                                                float* __restrict__ out) {
    const int lane = threadIdx.x & 31;
    const int wave = threadIdx.x >> 5;
    const int mt = blockIdx.x;
    const int nt = blockIdx.y * 8 + wave;        // 0..31
    const int row = mt * 16 + (lane & 15);
    const int kb  = (lane >> 4) * 16;
    v8f acc0 = {}, acc1 = {};
    #pragma unroll 2
    for (int kt = 0; kt < INNER / 32; kt += 2) {
        const _Float16* ap0 = oin + (size_t)row * INNER + kt * 32 + kb;
        const _Float16* ap1 = ap0 + 32;
        v16h a0 = cat16(*(const v8h*)ap0, *(const v8h*)(ap0 + 8));
        v16h a1 = cat16(*(const v8h*)ap1, *(const v8h*)(ap1 + 8));
        v16h b0 = *(const v16h*)(wp + (((size_t)kt * (DIM / 16) + nt) * 32 + lane) * 16);
        v16h b1 = *(const v16h*)(wp + (((size_t)(kt + 1) * (DIM / 16) + nt) * 32 + lane) * 16);
        acc0 = wmma_f16(a0, b0, acc0);
        acc1 = wmma_f16(a1, b1, acc1);
    }
    const int col = nt * 16 + (lane & 15);
    const float bias = bo[col];
    #pragma unroll
    for (int i = 0; i < 8; ++i) {
        const int r = mt * 16 + i + (lane >> 4) * 8;
        out[(size_t)r * DIM + col] = acc0[i] + acc1[i] + bias;
    }
}

// ---------------------------------------------------------------------------
extern "C" void kernel_launch(void* const* d_in, const int* in_sizes, int n_in,
                              void* d_out, int out_size, void* d_ws, size_t ws_size,
                              hipStream_t stream) {
    const float* x        = (const float*)d_in[0];
    const float* rel_bias = (const float*)d_in[1];
    const float* ln_scale = (const float*)d_in[2];
    const float* w_qkv    = (const float*)d_in[3];
    const float* w_out    = (const float*)d_in[4];
    const float* b_out    = (const float*)d_in[5];
    float* out = (float*)d_out;

    char* ws = (char*)d_ws;
    _Float16* xn_h   = (_Float16*)(ws);                       // 4096*512   f16
    _Float16* wqkv_p = (_Float16*)(ws + 4194304);             // 512*768    f16
    _Float16* wout_p = (_Float16*)(ws + 4980736);             // 256*512    f16
    _Float16* q_h    = (_Float16*)(ws + 5242880);             // 2*8*2048*32
    _Float16* k_h    = (_Float16*)(ws + 7340032);
    _Float16* vt_h   = (_Float16*)(ws + 9437184);
    _Float16* o_h    = (_Float16*)(ws + 11534336);            // 4096*256

    ln_kernel<<<BN, DIM, 0, stream>>>(x, ln_scale, xn_h);

    { // pack w_qkv: (512/32)*(768/16) = 768 tiles * 32 lanes
        const int threads = 768 * 32;
        pack_b<<<(threads + 255) / 256, 256, 0, stream>>>(w_qkv, wqkv_p, DIM, NQKV);
    }
    { // pack w_out: (256/32)*(512/16) = 256 tiles * 32 lanes
        const int threads = 256 * 32;
        pack_b<<<(threads + 255) / 256, 256, 0, stream>>>(w_out, wout_p, INNER, DIM);
    }

    qkv_gemm<<<dim3(BN / 16, NQKV / 128), 256, 0, stream>>>(xn_h, wqkv_p, q_h, k_h, vt_h);

    attn_kernel<<<dim3(SEQ / 64, HEADS), 128, 0, stream>>>(q_h, k_h, vt_h, rel_bias, o_h);

    out_gemm<<<dim3(BN / 16, DIM / 128), 256, 0, stream>>>(o_h, wout_p, b_out, out);
}
